// CCellBaseLSTM_45999099740636
// MI455X (gfx1250) — compile-verified, matrix operand order checked by reading, not run on hardware
//
#include <hip/hip_runtime.h>

// Problem constants (match reference): T=2048, B=64, I=512, H=512, 4H=2048
#define TSEQ  2048
#define BAT   64
#define IIN   512
#define HID   512
#define GATES 2048
#define NT_G  (GATES / 16)   // 128 column tiles across all gates
#define KT_TOT 32            // 32 k-tiles of 32 over combined K=[x|h]=1024

typedef __attribute__((ext_vector_type(16))) __bf16 v16bf;
typedef __attribute__((ext_vector_type(8)))  __bf16 v8bf;
typedef __attribute__((ext_vector_type(8)))  float  v8f;

union ABlock { v16bf v; v8bf h[2]; };

__device__ __forceinline__ float fsig(float x)  { return 1.0f / (1.0f + __expf(-x)); }
__device__ __forceinline__ float ftanh(float x) { return 1.0f - 2.0f / (__expf(2.0f * x) + 1.0f); }

// ---------------------------------------------------------------------------
// Pre-pass 1: swizzle fp32 W[N=4H][K] (row-major) into the bf16 WMMA
// B-fragment layout.  Fragment (kt, nt): lane L holds column n = nt*16+L%16,
// elems j=0..7 -> K = kt*32 + (L/16)*8 + j, j=8..15 -> +16 further.
// Fragments stored kt-major: frag = kt*NT + nt, 512 bf16 (1 KB) contiguous.
// w_ih occupies kt 0..15 of the combined buffer, w_hh kt 16..31 (contiguous).
// ---------------------------------------------------------------------------
__global__ __launch_bounds__(256) void swizzle_w_kernel(
    const float* __restrict__ W, __bf16* __restrict__ Wsw, int N, int K) {
  int tid = blockIdx.x * blockDim.x + threadIdx.x;   // one thread = 16 bf16
  int total = (N * K) >> 4;
  if (tid >= total) return;
  int lane = tid & 31;
  int frag = tid >> 5;
  int NT = N >> 4;
  int kt = frag / NT;
  int nt = frag % NT;
  int n  = nt * 16 + (lane & 15);
  int ko = (lane >> 4) << 3;
  const float* src = W + (size_t)n * K + kt * 32 + ko;
  __bf16* dst = Wsw + (size_t)frag * 512 + lane * 16;
  v8bf lo, hi;
#pragma unroll
  for (int j = 0; j < 8; ++j) { lo[j] = (__bf16)src[j]; hi[j] = (__bf16)src[16 + j]; }
  *(v8bf*)(dst)     = lo;
  *(v8bf*)(dst + 8) = hi;
}

// Pre-pass 2: bias = b_ih + b_hh ; zero h0 (bf16) and c0 (f32)
__global__ __launch_bounds__(256) void init_misc_kernel(
    const float* __restrict__ b_ih, const float* __restrict__ b_hh,
    float* __restrict__ bias, __bf16* __restrict__ h0, float* __restrict__ c0) {
  int i = blockIdx.x * blockDim.x + threadIdx.x;
  if (i < GATES) bias[i] = b_ih[i] + b_hh[i];
  if (i < BAT * HID) { h0[i] = (__bf16)0.0f; c0[i] = 0.0f; }
}

// ---------------------------------------------------------------------------
// One LSTM timestep.  grid = 16 WGs (32 hidden cols each) x 256 thr (8 waves).
// Wave (mt, nth) owns the 16(batch)x16(hidden) tile and accumulates all four
// gates.  Combined K = [x_t | h_{t-1}] = 1024 = 16 chunks x 64 K.
// Software pipeline:
//   - B fragments: 3-slot rotating register buffer, loads issued 2 kt ahead.
//   - A fragments: both kt-halves of a chunk loaded once per chunk
//     (one s_wait_dscnt per chunk).
//   - A panel: LDS double-buffered one chunk ahead (global->regs in half 0,
//     regs->LDS in half 1, single barrier per chunk).
// ---------------------------------------------------------------------------
__global__ __launch_bounds__(256) void lstm_step_kernel(
    const float*  __restrict__ x,        // [T,B,I] fp32
    const __bf16* __restrict__ wsw,      // combined swizzled weights, 32 kt-tiles
    const float*  __restrict__ bias,     // [4H]
    const __bf16* __restrict__ h_in,     // [B,H] bf16 (previous h)
    __bf16*       __restrict__ h_out,    // [B,H] bf16 (next h)
    float*        __restrict__ c,        // [B,H] f32 state
    float*        __restrict__ out,      // [T,B,H] f32
    float*        __restrict__ hc_final, // tail of d_out: h then c
    int t) {
  // two 64x64 bf16 chunks; rows padded to 80 elems (160 B) -> 32B-aligned
  // vector accesses + bank staggering.  2*64*80*2 = 20 KB LDS.
  __shared__ __align__(32) __bf16 As[2][64][80];

  const int tid  = threadIdx.x;
  const int lane = tid & 31;
  const int wave = tid >> 5;
  const int mt   = wave & 3;                 // batch-row tile 0..3
  const int nth  = wave >> 2;                // hidden-col tile 0..1
  const int hcol = blockIdx.x * 32 + nth * 16;
  const int nb   = lane & 15;                // column within tile
  const int ko   = (lane >> 4) << 3;         // 0 or 8 (K-octet select)
  const int ntg0 = blockIdx.x * 2 + nth;     // col-tile index within one gate

  // per-lane base of B fragments: + g*16384 elems per gate, + kt*65536 per kt
  const __bf16* p0 = wsw + (size_t)ntg0 * 512 + lane * 16;

  // hoisted bias loads (scalar per lane, all 8 acc elems share the column)
  const float bi = bias[0 * HID + hcol + nb];
  const float bf = bias[1 * HID + hcol + nb];
  const float bg = bias[2 * HID + hcol + nb];
  const float bo = bias[3 * HID + hcol + nb];

  v8f acc[4];
#pragma unroll
  for (int g = 0; g < 4; ++g)
#pragma unroll
    for (int e = 0; e < 8; ++e) acc[g][e] = 0.0f;

  const int srow = tid >> 2;                 // staging: row 0..63
  const int scol = (tid & 3) * 16;           // staging: 16-col group
  const float* xrow = x + ((size_t)t * BAT + srow) * IIN + scol;
  const __bf16* hrow = h_in + (size_t)srow * HID + scol;

  // ---- prologue: stage chunk 0 (x), preload B fragments for kt=0,1 ----
  {
    v16bf vv;
#pragma unroll
    for (int j = 0; j < 16; j += 4) {
      float4 f = *(const float4*)(xrow + j);
      vv[j + 0] = (__bf16)f.x; vv[j + 1] = (__bf16)f.y;
      vv[j + 2] = (__bf16)f.z; vv[j + 3] = (__bf16)f.w;
    }
    *(v16bf*)&As[0][srow][scol] = vv;
  }
  v16bf bq[3][4];                            // rotating B pipeline, depth 2
#pragma unroll
  for (int g = 0; g < 4; ++g) {
    bq[0][g] = *(const v16bf*)(p0 + g * 16384);
    bq[1][g] = *(const v16bf*)(p0 + (size_t)65536 + g * 16384);
  }
  __syncthreads();

  float4 sreg0, sreg1, sreg2, sreg3;         // staged x (half 0 -> half 1)
  v16bf  hreg;                               // staged h

#pragma unroll
  for (int kc = 0; kc < 16; ++kc) {          // 16 chunks of 64 K
    const int buf = kc & 1;
    const int nkc = kc + 1;                  // chunk being staged
    const int kt0 = kc * 2;

    // both A fragments for this chunk (ISA 16-bit A layout); single dscnt
    // wait covers the whole chunk's 8 WMMAs.
    ABlock a0, a1;
    const __bf16* arow = &As[buf][mt * 16 + nb][ko];
    a0.h[0] = *(const v8bf*)(arow);
    a0.h[1] = *(const v8bf*)(arow + 16);
    a1.h[0] = *(const v8bf*)(arow + 32);
    a1.h[1] = *(const v8bf*)(arow + 48);

    // ---------------- half 0 (kt = kt0) ----------------
    if (kt0 + 2 < KT_TOT) {                  // B loads 2 kt ahead
#pragma unroll
      for (int g = 0; g < 4; ++g)
        bq[(kt0 + 2) % 3][g] =
            *(const v16bf*)(p0 + (size_t)(kt0 + 2) * 65536 + g * 16384);
    }
    // issue next chunk's global loads into registers
    if (nkc < 8) {
      const float* src = xrow + nkc * 64;
      sreg0 = *(const float4*)(src + 0);
      sreg1 = *(const float4*)(src + 4);
      sreg2 = *(const float4*)(src + 8);
      sreg3 = *(const float4*)(src + 12);
    } else if (nkc < 16) {
      hreg = *(const v16bf*)(hrow + (nkc - 8) * 64);
    }
    {
      const int s = kt0 % 3;
      acc[0] = __builtin_amdgcn_wmma_f32_16x16x32_bf16(
          false, a0.v, false, bq[s][0], (short)0, acc[0], false, false);
      acc[1] = __builtin_amdgcn_wmma_f32_16x16x32_bf16(
          false, a0.v, false, bq[s][1], (short)0, acc[1], true, false);
      acc[2] = __builtin_amdgcn_wmma_f32_16x16x32_bf16(
          false, a0.v, false, bq[s][2], (short)0, acc[2], true, false);
      acc[3] = __builtin_amdgcn_wmma_f32_16x16x32_bf16(
          false, a0.v, false, bq[s][3], (short)0, acc[3], true, false);
    }

    // ---------------- half 1 (kt = kt0 + 1) ----------------
    if (kt0 + 3 < KT_TOT) {
#pragma unroll
      for (int g = 0; g < 4; ++g)
        bq[(kt0 + 3) % 3][g] =
            *(const v16bf*)(p0 + (size_t)(kt0 + 3) * 65536 + g * 16384);
    }
    // commit staged data to the other LDS buffer
    if (nkc < 16) {
      v16bf vv;
      if (nkc < 8) {
        vv[0]  = (__bf16)sreg0.x; vv[1]  = (__bf16)sreg0.y;
        vv[2]  = (__bf16)sreg0.z; vv[3]  = (__bf16)sreg0.w;
        vv[4]  = (__bf16)sreg1.x; vv[5]  = (__bf16)sreg1.y;
        vv[6]  = (__bf16)sreg1.z; vv[7]  = (__bf16)sreg1.w;
        vv[8]  = (__bf16)sreg2.x; vv[9]  = (__bf16)sreg2.y;
        vv[10] = (__bf16)sreg2.z; vv[11] = (__bf16)sreg2.w;
        vv[12] = (__bf16)sreg3.x; vv[13] = (__bf16)sreg3.y;
        vv[14] = (__bf16)sreg3.z; vv[15] = (__bf16)sreg3.w;
      } else {
        vv = hreg;
      }
      *(v16bf*)&As[buf ^ 1][srow][scol] = vv;
    }
    {
      const int s = (kt0 + 1) % 3;
      acc[0] = __builtin_amdgcn_wmma_f32_16x16x32_bf16(
          false, a1.v, false, bq[s][0], (short)0, acc[0], false, false);
      acc[1] = __builtin_amdgcn_wmma_f32_16x16x32_bf16(
          false, a1.v, false, bq[s][1], (short)0, acc[1], true, false);
      acc[2] = __builtin_amdgcn_wmma_f32_16x16x32_bf16(
          false, a1.v, false, bq[s][2], (short)0, acc[2], true, false);
      acc[3] = __builtin_amdgcn_wmma_f32_16x16x32_bf16(
          false, a1.v, false, bq[s][3], (short)0, acc[3], true, false);
    }

    // one barrier per chunk, after the last read of the buffer that the
    // next chunk-store will overwrite
    if (kc < 15) __syncthreads();
  }

  // ---- fused LSTM pointwise epilogue ----
  // C-layout: VGPR e, lane L -> row m = mt*16 + e + 8*(L>=16), col = L%16.
  const int mbase = mt * 16 + ((lane >> 4) << 3);
  const int hidx  = hcol + nb;
#pragma unroll
  for (int e = 0; e < 8; ++e) {
    const float iv = fsig (acc[0][e] + bi);
    const float fv = fsig (acc[1][e] + bf);
    const float gv = ftanh(acc[2][e] + bg);
    const float ov = fsig (acc[3][e] + bo);
    const int m = mbase + e;
    const size_t ci = (size_t)m * HID + hidx;
    const float cnew = fv * c[ci] + iv * gv;
    const float hv = ov * ftanh(cnew);
    c[ci] = cnew;
    out[((size_t)t * BAT + m) * HID + hidx] = hv;
    h_out[ci] = (__bf16)hv;
    if (t == TSEQ - 1) {
      hc_final[ci] = hv;                       // final h
      hc_final[(size_t)BAT * HID + ci] = cnew; // final c
    }
  }
}

// ---------------------------------------------------------------------------
extern "C" void kernel_launch(void* const* d_in, const int* in_sizes, int n_in,
                              void* d_out, int out_size, void* d_ws, size_t ws_size,
                              hipStream_t stream) {
  (void)in_sizes; (void)n_in; (void)out_size; (void)ws_size;
  const float* x    = (const float*)d_in[0];   // word_seq [T,B,I]
  const float* w_ih = (const float*)d_in[1];   // [4H,I]
  const float* w_hh = (const float*)d_in[2];   // [4H,H]
  const float* b_ih = (const float*)d_in[3];
  const float* b_hh = (const float*)d_in[4];
  float* out = (float*)d_out;                  // [T,B,H] ++ h[B,H] ++ c[B,H]

  // Workspace layout (~4.3 MB total).  wih (kt 0..15) and whh (kt 16..31)
  // form ONE contiguous swizzled weight array of 32 kt-tiles (2 MiB each).
  char* ws = (char*)d_ws;
  __bf16* wsw    = (__bf16*)(ws);                               // 4 MiB
  __bf16* whh_sw = wsw + (size_t)16 * NT_G * 512;               // = ws + 2 MiB
  float*  bias   = (float*) (ws + (4u << 20));                  // 8 KiB
  __bf16* hbuf0  = (__bf16*)(ws + (4u << 20) + (16u << 10));    // 64 KiB
  __bf16* hbuf1  = hbuf0 + (size_t)BAT * HID;                   // 64 KiB
  float*  cbuf   = (float*)(hbuf1 + (size_t)BAT * HID);         // 128 KiB

  const int chunks = GATES * IIN / 16;                          // 65536
  swizzle_w_kernel<<<(chunks + 255) / 256, 256, 0, stream>>>(w_ih, wsw,    GATES, IIN);
  swizzle_w_kernel<<<(chunks + 255) / 256, 256, 0, stream>>>(w_hh, whh_sw, GATES, HID);
  init_misc_kernel<<<(BAT * HID + 255) / 256, 256, 0, stream>>>(b_ih, b_hh, bias, hbuf0, cbuf);

  float* hc_final = out + (size_t)TSEQ * BAT * HID;
  for (int t = 0; t < TSEQ; ++t) {
    const __bf16* hin = (t & 1) ? hbuf1 : hbuf0;   // ping-pong h state
    __bf16*      hout = (t & 1) ? hbuf0 : hbuf1;
    lstm_step_kernel<<<16, 256, 0, stream>>>(
        x, wsw, bias, hin, hout, cbuf, out, hc_final, t);
  }
}
